// GRU_16295105921625
// MI455X (gfx1250) — matrix operand order repaired and checked by reading, not compiled, not run on hardware
//
#include <hip/hip_runtime.h>

// Problem constants (from reference): B=512, H=1024, D=4096, P=2048, T=48
#define BB 512
#define HH 1024
#define DD 4096
#define PP 2048
#define TT 48

typedef __attribute__((ext_vector_type(16))) __bf16       v16bf;
typedef __attribute__((ext_vector_type(8)))  float        v8f;
typedef __attribute__((ext_vector_type(4)))  unsigned int u32x4;

union Frag {
  u32x4 q[2];   // 2 x 16B = 16 bf16
  v16bf v;
};

__device__ __forceinline__ unsigned short f2bf(float f) {
  unsigned u = __builtin_bit_cast(unsigned, f);
  unsigned r = u + 0x7FFFu + ((u >> 16) & 1u);   // round-to-nearest-even
  return (unsigned short)(r >> 16);
}

__device__ __forceinline__ float sigf(float x) {
  return 1.0f / (1.0f + __expf(-x));
}

__global__ void k_cvt(const float* __restrict__ in, unsigned short* __restrict__ out, int n) {
  int i = blockIdx.x * blockDim.x + threadIdx.x;
  if (i < n) out[i] = f2bf(in[i]);
}

// ---- CDNA5 async global->LDS copy (ASYNCcnt-tracked, no VGPR round trip) ----
// LDS operand: generic LDS pointer truncates to addr[31:0] == workgroup LDS offset.
__device__ __forceinline__ void async_ld16(const unsigned short* gptr, unsigned short* lptr) {
  unsigned lds_off = (unsigned)(unsigned long long)(uintptr_t)lptr;
  unsigned long long ga = (unsigned long long)(uintptr_t)gptr;
  asm volatile("global_load_async_to_lds_b128 %0, %1, off"
               :: "v"(lds_off), "v"(ga)
               : "memory");
}
__device__ __forceinline__ void wait_async0() {
  asm volatile("s_wait_asynccnt 0" ::: "memory");
}

// ---- tile geometry -------------------------------------------------------
// LDS tiles: row stride 72 bf16 (144B: 16B aligned, bank-conflict pad)
#define LDS_STRIDE 72

// A fragment (16x32 bf16, ISA layout): lanes 0-15 -> M=lane, K {0..7,16..23};
// lanes 16-31 -> M=lane-16, K {8..15,24..31}
__device__ __forceinline__ Frag load_a(const unsigned short* lds, int mbase, int kk, int lane) {
  Frag a;
  int m   = mbase + (lane & 15);
  int off = (lane < 16) ? 0 : 8;
  const unsigned short* p = &lds[m * LDS_STRIDE + kk + off];
  a.q[0] = *reinterpret_cast<const u32x4*>(p);
  a.q[1] = *reinterpret_cast<const u32x4*>(p + 16);
  return a;
}

// B fragment (32x16 = K x N): lane<16 -> N=lane, K {0..7,16..23}; lane>=16 -> N=lane-16, K {8..15,24..31}.
// B[k][n] = Wtile[n][k] -> two 16-byte runs 32 bytes apart from the staged row-major W tile.
__device__ __forceinline__ Frag load_b_lds(const unsigned short* ldsW, int rowbase, int kk, int lane) {
  Frag b;
  int row = rowbase + (lane & 15);
  int kc  = kk + ((lane < 16) ? 0 : 8);
  const unsigned short* p = &ldsW[row * LDS_STRIDE + kc];
  b.q[0] = *reinterpret_cast<const u32x4*>(p);
  b.q[1] = *reinterpret_cast<const u32x4*>(p + 16);
  return b;
}

// ---- fused GEMM + bias + sigmoid:  out = sigmoid(X[512,K] @ W[N,K]^T + bias) ----
__global__ __launch_bounds__(256) void k_outproj(
    const unsigned short* __restrict__ X,
    const unsigned short* __restrict__ W,
    const float* __restrict__ bias,
    float* __restrict__ outF, long strideF,
    unsigned short* __restrict__ outB,
    int N, int K) {
  __shared__ unsigned short ldsX[2][128 * LDS_STRIDE];  // 2 x 18.0 KB
  __shared__ unsigned short ldsW[2][64 * LDS_STRIDE];   // 2 x  9.0 KB
  int tid  = threadIdx.x;
  int lane = tid & 31;
  int wave = tid >> 5;
  int wm   = wave & 3;          // 4 waves along M (32 rows each)
  int wn   = wave >> 2;         // 2 waves along N (32 cols each)
  int m0   = blockIdx.y * 128;
  int n0   = blockIdx.x * 64;

  // async-stage one 64-wide K chunk of X (128x64) and W (64x64) into buffer `buf`
  auto stage = [&](int buf, int k0) {
#pragma unroll
    for (int j = 0; j < 4; ++j) {            // X: 1024 chunks / 256 thr
      int i = tid + 256 * j, row = i >> 3, cc = i & 7;
      async_ld16(X + (size_t)(m0 + row) * K + k0 + cc * 8,
                 &ldsX[buf][row * LDS_STRIDE + cc * 8]);
    }
#pragma unroll
    for (int j = 0; j < 2; ++j) {            // W: 512 chunks / 256 thr
      int i = tid + 256 * j, row = i >> 3, cc = i & 7;
      async_ld16(W + (size_t)(n0 + row) * K + k0 + cc * 8,
                 &ldsW[buf][row * LDS_STRIDE + cc * 8]);
    }
  };

  v8f zero = {};
  v8f acc[2][2];
#pragma unroll
  for (int i = 0; i < 2; ++i)
#pragma unroll
    for (int j = 0; j < 2; ++j) acc[i][j] = zero;

  int nc = K >> 6;
  stage(0, 0);
  for (int c = 0; c < nc; ++c) {
    int buf = c & 1;
    wait_async0();        // chunk c resident in LDS
    __syncthreads();
    if (c + 1 < nc) stage(buf ^ 1, (c + 1) << 6);   // overlap next chunk with compute

    // Load ALL fragments for this chunk first (one deep ds clause), then an
    // uninterrupted wmma train -> staggered s_wait_dscnt instead of hard 0-waits.
    Frag a[2][2], b[2][2];                   // [kk][ms] / [kk][ns]
#pragma unroll
    for (int kq = 0; kq < 2; ++kq) {
      a[kq][0] = load_a(ldsX[buf], wm * 32,      kq * 32, lane);
      a[kq][1] = load_a(ldsX[buf], wm * 32 + 16, kq * 32, lane);
      b[kq][0] = load_b_lds(ldsW[buf], wn * 32,      kq * 32, lane);
      b[kq][1] = load_b_lds(ldsW[buf], wn * 32 + 16, kq * 32, lane);
    }
#pragma unroll
    for (int kq = 0; kq < 2; ++kq)
#pragma unroll
      for (int ns = 0; ns < 2; ++ns) {
        acc[0][ns] = __builtin_amdgcn_wmma_f32_16x16x32_bf16(
            false, a[kq][0].v, false, b[kq][ns].v, (short)0, acc[0][ns], false, false);
        acc[1][ns] = __builtin_amdgcn_wmma_f32_16x16x32_bf16(
            false, a[kq][1].v, false, b[kq][ns].v, (short)0, acc[1][ns], false, false);
      }
  }

  int half = lane >> 4, ln = lane & 15;
#pragma unroll
  for (int ms = 0; ms < 2; ++ms)
#pragma unroll
    for (int ns = 0; ns < 2; ++ns) {
      int col  = n0 + wn * 32 + ns * 16 + ln;
      float bv = bias[col];
#pragma unroll
      for (int j = 0; j < 8; ++j) {
        int row   = m0 + wm * 32 + ms * 16 + half * 8 + j;
        float val = sigf(acc[ms][ns][j] + bv);
        outF[(size_t)row * strideF + col] = val;
        outB[(size_t)row * N + col]       = f2bf(val);
      }
    }
}

// ---- fused GRU gate GEMM:  h = (1-z) * tanh(...) with 3 gate accumulators ----
// W is [3*1024, K]; output cols n use W rows {n, n+1024, n+2048}; block covers 32 cols.
__global__ __launch_bounds__(256) void k_gate(
    const unsigned short* __restrict__ X,
    const unsigned short* __restrict__ W,
    const float* __restrict__ bih,
    const float* __restrict__ bhh,
    float* __restrict__ outF, long strideF,
    unsigned short* __restrict__ outB,
    int K) {
  __shared__ unsigned short ldsX[2][128 * LDS_STRIDE];  // 2 x 18.0 KB
  __shared__ unsigned short ldsW[2][96 * LDS_STRIDE];   // 2 x 13.5 KB (3 gates x 32 cols)
  int tid  = threadIdx.x;
  int lane = tid & 31;
  int wave = tid >> 5;
  int wm   = wave & 3;
  int wn   = wave >> 2;
  int m0   = blockIdx.y * 128;
  int nblk = blockIdx.x * 32;             // block's output-column base

  auto stage = [&](int buf, int k0) {
#pragma unroll
    for (int j = 0; j < 4; ++j) {            // X: 1024 chunks / 256 thr
      int i = tid + 256 * j, row = i >> 3, cc = i & 7;
      async_ld16(X + (size_t)(m0 + row) * K + k0 + cc * 8,
                 &ldsX[buf][row * LDS_STRIDE + cc * 8]);
    }
#pragma unroll
    for (int j = 0; j < 3; ++j) {            // W: 96 rows x 8 chunks = 768 / 256 thr
      int i = tid + 256 * j, trow = i >> 3, cc = i & 7;
      int g = trow >> 5, c5 = trow & 31;     // tile row -> gate, col-in-block
      async_ld16(W + (size_t)(g * HH + nblk + c5) * K + k0 + cc * 8,
                 &ldsW[buf][trow * LDS_STRIDE + cc * 8]);
    }
  };

  v8f zero = {};
  v8f acc[3][2];
#pragma unroll
  for (int g = 0; g < 3; ++g)
#pragma unroll
    for (int ms = 0; ms < 2; ++ms) acc[g][ms] = zero;

  int nc = K >> 6;
  stage(0, 0);
  for (int c = 0; c < nc; ++c) {
    int buf = c & 1;
    wait_async0();
    __syncthreads();
    if (c + 1 < nc) stage(buf ^ 1, (c + 1) << 6);

    Frag a[2][2], b[2][3];                   // [kk][ms] / [kk][gate]
#pragma unroll
    for (int kq = 0; kq < 2; ++kq) {
      a[kq][0] = load_a(ldsX[buf], wm * 32,      kq * 32, lane);
      a[kq][1] = load_a(ldsX[buf], wm * 32 + 16, kq * 32, lane);
#pragma unroll
      for (int g = 0; g < 3; ++g)
        b[kq][g] = load_b_lds(ldsW[buf], g * 32 + wn * 16, kq * 32, lane);
    }
#pragma unroll
    for (int kq = 0; kq < 2; ++kq)
#pragma unroll
      for (int g = 0; g < 3; ++g) {
        acc[g][0] = __builtin_amdgcn_wmma_f32_16x16x32_bf16(
            false, a[kq][0].v, false, b[kq][g].v, (short)0, acc[g][0], false, false);
        acc[g][1] = __builtin_amdgcn_wmma_f32_16x16x32_bf16(
            false, a[kq][1].v, false, b[kq][g].v, (short)0, acc[g][1], false, false);
      }
  }

  int half = lane >> 4, ln = lane & 15;
  int col  = nblk + wn * 16 + ln;
  float br  = bih[col]          + bhh[col];
  float bz  = bih[HH + col]     + bhh[HH + col];
  float bni = bih[2 * HH + col];
  float bnh = bhh[2 * HH + col];
#pragma unroll
  for (int ms = 0; ms < 2; ++ms)
#pragma unroll
    for (int j = 0; j < 8; ++j) {
      int row  = m0 + wm * 32 + ms * 16 + half * 8 + j;
      float r  = sigf(acc[0][ms][j] + br);
      float z  = sigf(acc[1][ms][j] + bz);
      float nn = tanhf(acc[2][ms][j] + bni + r * bnh);
      float h  = (1.0f - z) * nn;
      outF[(size_t)row * strideF + col] = h;
      outB[(size_t)row * HH + col]      = f2bf(h);
    }
}

// ---- host orchestration -------------------------------------------------
extern "C" void kernel_launch(void* const* d_in, const int* in_sizes, int n_in,
                              void* d_out, int out_size, void* d_ws, size_t ws_size,
                              hipStream_t stream) {
  (void)in_sizes; (void)n_in; (void)out_size; (void)ws_size;

  const float* noise = (const float*)d_in[0];
  const float* dWih  = (const float*)d_in[1];
  // d_in[2] (d_Whh) and d_in[6] (p_Whh) are dead: h==0 in the reference.
  const float* dbih  = (const float*)d_in[3];
  const float* dbhh  = (const float*)d_in[4];
  const float* pWih  = (const float*)d_in[5];
  const float* pbih  = (const float*)d_in[7];
  const float* pbhh  = (const float*)d_in[8];
  const float* dWout = (const float*)d_in[9];
  const float* dbout = (const float*)d_in[10];
  const float* pWout = (const float*)d_in[11];
  const float* pbout = (const float*)d_in[12];

  float* ds = (float*)d_out;                       // [B,T,D]
  float* ps = ds + (size_t)BB * TT * DD;           // [B,T,P]
  float* dh = ps + (size_t)BB * TT * PP;           // [B,T,H]
  float* ph = dh + (size_t)BB * TT * HH;           // [B,T,H]

  char* wsp = (char*)d_ws;
  size_t off = 0;
  auto take = [&](size_t elems) -> unsigned short* {
    unsigned short* p = (unsigned short*)(wsp + off);
    off += (elems * 2 + 255) & ~(size_t)255;
    return p;
  };
  unsigned short* bWih_d  = take((size_t)3 * HH * DD);  // 25.2 MB
  unsigned short* bWih_p  = take((size_t)3 * HH * PP);  // 12.6 MB
  unsigned short* bWout_d = take((size_t)DD * HH);      //  8.4 MB
  unsigned short* bWout_p = take((size_t)PP * HH);      //  4.2 MB
  unsigned short* bNoise  = take((size_t)BB * HH);
  unsigned short* bXd     = take((size_t)BB * DD);
  unsigned short* bXp     = take((size_t)BB * PP);
  unsigned short* bHd     = take((size_t)BB * HH);
  unsigned short* bHp     = take((size_t)BB * HH);

  auto cvt = [&](const float* in, unsigned short* out, size_t n) {
    k_cvt<<<dim3((unsigned)((n + 255) / 256)), dim3(256), 0, stream>>>(in, out, (int)n);
  };
  cvt(dWih,  bWih_d,  (size_t)3 * HH * DD);
  cvt(pWih,  bWih_p,  (size_t)3 * HH * PP);
  cvt(dWout, bWout_d, (size_t)DD * HH);
  cvt(pWout, bWout_p, (size_t)PP * HH);
  cvt(noise, bNoise,  (size_t)BB * HH);

  dim3 blk(256);
  dim3 gD(DD / 64, BB / 128);   // 64 x 4
  dim3 gP(PP / 64, BB / 128);   // 32 x 4
  dim3 gG(HH / 32, BB / 128);   // 32 x 4

  // t=0 samples: dc0 = sigmoid(noise @ Wout^T + b)
  k_outproj<<<gD, blk, 0, stream>>>(bNoise, bWout_d, dbout, ds, (long)TT * DD, bXd, DD, HH);
  k_outproj<<<gP, blk, 0, stream>>>(bNoise, bWout_p, pbout, ps, (long)TT * PP, bXp, PP, HH);

  for (int t = 0; t < TT; ++t) {
    k_gate<<<gG, blk, 0, stream>>>(bXd, bWih_d, dbih, dbhh,
                                   dh + (size_t)t * HH, (long)TT * HH, bHd, DD);
    k_gate<<<gG, blk, 0, stream>>>(bXp, bWih_p, pbih, pbhh,
                                   ph + (size_t)t * HH, (long)TT * HH, bHp, PP);
    if (t + 1 < TT) {
      k_outproj<<<gD, blk, 0, stream>>>(bHd, bWout_d, dbout,
                                        ds + (size_t)(t + 1) * DD, (long)TT * DD, bXd, DD, HH);
      k_outproj<<<gP, blk, 0, stream>>>(bHp, bWout_p, pbout,
                                        ps + (size_t)(t + 1) * PP, (long)TT * PP, bXp, PP, HH);
    }
  }
}
